// PointNet2Discriminator_27290222199384
// MI455X (gfx1250) — matrix operand order repaired and checked by reading, compiled, not run on hardware
//
#include <hip/hip_runtime.h>

// ---------------------------------------------------------------------------
// PointNet++ discriminator for MI455X (gfx1250, wave32, WMMA).
// GEMM-heavy SA MLPs run on v_wmma_f32_16x16x32_bf16; irregular stages
// (FPS / ball query / attention / head MLPs) are latency-bound scalar kernels.
// ---------------------------------------------------------------------------

#define B_SZ 4
#define GRID_G 31
#define N0_PTS (GRID_G * GRID_G * GRID_G) // 29791
#define HID 256

#define CDIV(a, b) (((a) + (b) - 1) / (b))

typedef __attribute__((ext_vector_type(16))) __bf16 v16bf;
typedef __attribute__((ext_vector_type(8))) float v8f;

__device__ __forceinline__ unsigned short f2bf(float f) {
  union { float f; unsigned u; } x; x.f = f;
  unsigned u = x.u;
  u += 0x7FFFu + ((u >> 16) & 1u);       // round-to-nearest-even
  return (unsigned short)(u >> 16);
}
__device__ __forceinline__ float bf2f(unsigned short h) {
  union { unsigned u; float f; } x; x.u = ((unsigned)h) << 16;
  return x.f;
}

// ---------------------------------------------------------------------------
// Voxel grid -> point features (closed-form: mean=15*vl, min=0, max=30*vl)
// data0 = [dec(3), norm(3), color(3)] ; coords = dec
// ---------------------------------------------------------------------------
__global__ void ply2data_kernel(const float* __restrict__ neigh,
                                const float* __restrict__ vlp,
                                float* __restrict__ coords0,
                                unsigned short* __restrict__ feats0, int total) {
  int t = blockIdx.x * blockDim.x + threadIdx.x;
  if (t >= total) return;
  int b = t / N0_PTS, n = t % N0_PTS;
  int i = n / (GRID_G * GRID_G);
  int rem = n % (GRID_G * GRID_G);
  int j = rem / GRID_G, k = rem % GRID_G;
  float vl = vlp[0];
  float dx = (i - 15) * vl, dy = (j - 15) * vl, dz = (k - 15) * vl;
  float nx = i / 30.0f, ny = j / 30.0f, nz = k / 30.0f;
  const float* base = neigh + (size_t)b * 3 * N0_PTS;
  float occ = base[n], per = base[N0_PTS + n], vis = base[2 * N0_PTS + n];
  float cr = 0.f, cg = 0.f, cb = 0.f;
  if (occ != 0.f) { cr = cg = cb = 128.0f / 255.0f; }
  if (per != 0.f) { cr = 1.f; cg = 1.f; cb = 0.f; }
  if (vis > 0.f)  { cr = 0.f; cg = vis; cb = 0.f; }
  coords0[t * 3 + 0] = dx; coords0[t * 3 + 1] = dy; coords0[t * 3 + 2] = dz;
  unsigned short* f = feats0 + (size_t)t * 9;
  f[0] = f2bf(dx); f[1] = f2bf(dy); f[2] = f2bf(dz);
  f[3] = f2bf(nx); f[4] = f2bf(ny); f[5] = f2bf(nz);
  f[6] = f2bf(cr); f[7] = f2bf(cg); f[8] = f2bf(cb);
}

// ---------------------------------------------------------------------------
// Farthest point sampling: one 1024-thread block per batch (serial argmax chain)
// ---------------------------------------------------------------------------
__global__ void __launch_bounds__(1024) fps_kernel(const float* __restrict__ coords,
                                                   float* __restrict__ dists,
                                                   int* __restrict__ fpsIdx,
                                                   int N, int npoint) {
  __shared__ float sv[1024];
  __shared__ int si[1024];
  __shared__ int sLast;
  int b = blockIdx.x, tid = threadIdx.x;
  const float* xyz = coords + (size_t)b * N * 3;
  float* dd = dists + (size_t)b * N;
  for (int i = tid; i < N; i += 1024) dd[i] = 1e10f;
  if (tid == 0) { fpsIdx[b * npoint] = 0; sLast = 0; }
  __syncthreads();
  for (int t = 1; t < npoint; ++t) {
    int last = sLast;
    float lx = xyz[last * 3], ly = xyz[last * 3 + 1], lz = xyz[last * 3 + 2];
    float best = -1.0f; int bi = 0;
    for (int i = tid; i < N; i += 1024) {
      float dx = xyz[i * 3] - lx, dy = xyz[i * 3 + 1] - ly, dz = xyz[i * 3 + 2] - lz;
      float d = dx * dx + dy * dy + dz * dz;
      float old = dd[i];
      d = d < old ? d : old;
      dd[i] = d;
      if (d > best) { best = d; bi = i; }
    }
    sv[tid] = best; si[tid] = bi;
    __syncthreads();
    for (int off = 512; off > 0; off >>= 1) {
      if (tid < off) {
        float ov = sv[tid + off]; int oi = si[tid + off];
        if (ov > sv[tid] || (ov == sv[tid] && oi < si[tid])) { sv[tid] = ov; si[tid] = oi; }
      }
      __syncthreads();
    }
    if (tid == 0) { sLast = si[0]; fpsIdx[b * npoint + t] = si[0]; }
    __syncthreads();
  }
}

__global__ void gather_xyz_kernel(const float* __restrict__ coords,
                                  const int* __restrict__ fpsIdx,
                                  float* __restrict__ newXyz,
                                  int total, int npoint, int N) {
  int t = blockIdx.x * blockDim.x + threadIdx.x;
  if (t >= total) return;
  int b = t / npoint;
  int idx = fpsIdx[t];
  const float* src = coords + ((size_t)b * N + idx) * 3;
  newXyz[t * 3 + 0] = src[0];
  newXyz[t * 3 + 1] = src[1];
  newXyz[t * 3 + 2] = src[2];
}

// First 32 in-radius indices in index order (matches top_k-on-indices trick)
__global__ void ball_query_kernel(const float* __restrict__ coords,
                                  const float* __restrict__ newXyz,
                                  int* __restrict__ gi,
                                  int total, int npoint, int N, float r2) {
  int t = blockIdx.x * blockDim.x + threadIdx.x;
  if (t >= total) return;
  int b = t / npoint;
  const float* xyz = coords + (size_t)b * N * 3;
  float cx = newXyz[t * 3], cy = newXyz[t * 3 + 1], cz = newXyz[t * 3 + 2];
  int* dst = gi + (size_t)t * 32;
  int cnt = 0, first = 0;
  for (int j = 0; j < N; ++j) {
    float dx = xyz[j * 3] - cx, dy = xyz[j * 3 + 1] - cy, dz = xyz[j * 3 + 2] - cz;
    float sq = dx * dx + dy * dy + dz * dz;
    if (sq <= r2) {
      if (cnt == 0) first = j;
      dst[cnt++] = j;
      if (cnt == 32) break;
    }
  }
  for (int k = cnt; k < 32; ++k) dst[k] = first;
}

// Build grouped GEMM input X (M x Kpad, bf16): [xyz-center | feats | 0-pad]
__global__ void group_kernel(const float* __restrict__ coords,
                             const float* __restrict__ newXyz,
                             const int* __restrict__ gi,
                             const unsigned short* __restrict__ feats,
                             unsigned short* __restrict__ X,
                             int total, int npoint, int Kpad, int C, int N) {
  int t = blockIdx.x * blockDim.x + threadIdx.x;
  if (t >= total) return;
  int c = t % Kpad;
  int row = t / Kpad;            // (b*npoint + q)*32 + s
  int bq = row >> 5;             // b*npoint + q
  int b = bq / npoint;
  int idx = gi[row];
  unsigned short val;
  if (c < 3) {
    float v = coords[((size_t)b * N + idx) * 3 + c] - newXyz[(size_t)bq * 3 + c];
    val = f2bf(v);
  } else if (c < 3 + C) {
    val = feats[((size_t)b * N + idx) * C + (c - 3)];
  } else {
    val = 0;
  }
  X[(size_t)row * Kpad + c] = val;
}

// ---------------------------------------------------------------------------
// bf16 WMMA GEMM: Out = relu(A @ W + bias), A (MxK), W (KxN), K,N mult 32,
// M mult 64. Block = 256 thr = 8 waves; tile 64x32; each wave a 16x16 tile.
// LDS tiles use a 40-halfword row stride: keeps 16 B alignment for b128
// merges while stepping 20 banks per lane (gcd(20,64)=4 -> conflict-free).
// B tile is staged TRANSPOSED (N-major) so the B fragment is two contiguous
// 16 B reads per lane (ds_load_b128), same as the A fragment.
// ---------------------------------------------------------------------------
#define ASTR 40
#define BSTR 40

__global__ void __launch_bounds__(256) gemm_bf16_wmma_kernel(
    const unsigned short* __restrict__ A, const unsigned short* __restrict__ Bw,
    const float* __restrict__ bias, unsigned short* __restrict__ Out,
    int M, int N, int K) {
  __shared__ alignas(16) unsigned short As[64 * ASTR];
  __shared__ alignas(16) unsigned short Bs[32 * BSTR];
  const int tid = threadIdx.x;
  const int wave = tid >> 5, lane = tid & 31;
  const int wm = wave >> 1, wn = wave & 1;
  const int m0 = blockIdx.x * 64, n0 = blockIdx.y * 32;
  const int lhalf = lane >> 4, l16 = lane & 15;

  union { unsigned int u[8]; v16bf v; } afrag, bfrag;
  union { v8f v; float f[8]; } acc;
#pragma unroll
  for (int i = 0; i < 8; ++i) acc.f[i] = 0.0f;

  // per-thread staging coordinates
  const int arow = tid >> 2, aquad = tid & 3;   // A: 64 rows x 4 uint4
  const int bn = tid & 31, bkq = tid >> 5;      // B: col n, k-quad (4 k each)

  for (int kk = 0; kk < K; kk += 32) {
    { // stage A tile 64x32 (one uint4 per thread, coalesced)
      const uint4* src = (const uint4*)(A + (size_t)(m0 + arow) * K + kk + aquad * 8);
      *(uint4*)(&As[arow * ASTR + aquad * 8]) = *src;
    }
    { // stage B tile 32x32 transposed -> Bs[n][k]; 4 coalesced row-segment
      // u16 loads per thread, one aligned uint2 LDS store
      unsigned int e0 = Bw[(size_t)(kk + 4 * bkq + 0) * N + n0 + bn];
      unsigned int e1 = Bw[(size_t)(kk + 4 * bkq + 1) * N + n0 + bn];
      unsigned int e2 = Bw[(size_t)(kk + 4 * bkq + 2) * N + n0 + bn];
      unsigned int e3 = Bw[(size_t)(kk + 4 * bkq + 3) * N + n0 + bn];
      uint2 p;
      p.x = e0 | (e1 << 16);
      p.y = e2 | (e3 << 16);
      *(uint2*)(&Bs[bn * BSTR + 4 * bkq]) = p;
    }
    __syncthreads();
    { // A fragment: 16x32, interleaved K-pair layout (ISA 7.12.2)
      int mrow = wm * 16 + l16;
#pragma unroll
      for (int v = 0; v < 8; ++v) {
        int kO = (v < 4) ? (lhalf * 8 + 2 * v) : (16 + lhalf * 8 + 2 * (v - 4));
        afrag.u[v] = *(const unsigned int*)(&As[mrow * ASTR + kO]);
      }
    }
    { // B fragment: 32x16, interleaved K-pair layout (same as A); from Bs[n][k]
      int ncol = wn * 16 + l16;
#pragma unroll
      for (int v = 0; v < 8; ++v) {
        int kO = (v < 4) ? (lhalf * 8 + 2 * v) : (16 + lhalf * 8 + 2 * (v - 4));
        bfrag.u[v] = *(const unsigned int*)(&Bs[ncol * BSTR + kO]);
      }
    }
    acc.v = __builtin_amdgcn_wmma_f32_16x16x32_bf16(
        false, afrag.v, false, bfrag.v, (short)0, acc.v, false, false);
    __syncthreads();
  }
  int ncol = n0 + wn * 16 + l16;
  float bb = bias[ncol];
#pragma unroll
  for (int v = 0; v < 8; ++v) {
    int mrow = m0 + wm * 16 + v + 8 * lhalf;
    float val = acc.f[v] + bb;
    val = val > 0.f ? val : 0.f;
    Out[(size_t)mrow * N + ncol] = f2bf(val);
  }
}

// Max over the 32 samples per query point
__global__ void maxpool_kernel(const unsigned short* __restrict__ H,
                               unsigned short* __restrict__ F, int total, int C) {
  int t = blockIdx.x * blockDim.x + threadIdx.x;
  if (t >= total) return;
  int c = t % C, p = t / C;
  float m = -1e30f;
  for (int s = 0; s < 32; ++s) {
    float v = bf2f(H[((size_t)p * 32 + s) * C + c]);
    m = v > m ? v : m;
  }
  F[(size_t)p * C + c] = f2bf(m);
}

// Zero-pad f32 weight (KxN) to bf16 (Kpad x N)
__global__ void wpad_kernel(const float* __restrict__ W, unsigned short* __restrict__ dst,
                            int K, int N, int Kpad) {
  int t = blockIdx.x * blockDim.x + threadIdx.x;
  if (t >= Kpad * N) return;
  int k = t / N, n = t % N;
  dst[t] = (k < K) ? f2bf(W[(size_t)k * N + n]) : (unsigned short)0;
}

// ---------------------------------------------------------------------------
// Head kernels (tiny, latency-bound): mask ordering, gathers, naive linears,
// attention, reductions.
// ---------------------------------------------------------------------------
__global__ void mask_order_kernel(const int* __restrict__ mask, int* __restrict__ selfIdx,
                                  int* __restrict__ order, int Bn, int A) {
  int b = threadIdx.x;
  if (b >= Bn) return;
  int si = 0;
  for (int j = 0; j < A; ++j) if (mask[b * A + j] != 0) { si = j; break; }
  selfIdx[b] = si;
  int c = 0;
  for (int j = 0; j < A && c < A - 1; ++j) if (mask[b * A + j] == 0) order[b * (A - 1) + c++] = j;
  for (int j = 0; j < A && c < A - 1; ++j) if (mask[b * A + j] != 0) order[b * (A - 1) + c++] = j;
}

__global__ void gather_self_kernel(const float* __restrict__ snp, const int* __restrict__ selfIdx,
                                   float* __restrict__ selfP) {
  int t = blockIdx.x * blockDim.x + threadIdx.x;
  if (t >= B_SZ * 10) return;
  int b = t / 10, p = t % 10;
  selfP[t] = snp[(b * 17 + selfIdx[b]) * 10 + p];
}

__global__ void gather_neigh_kernel(const float* __restrict__ snp, const int* __restrict__ order,
                                    float* __restrict__ neighP) {
  int t = blockIdx.x * blockDim.x + threadIdx.x;
  if (t >= B_SZ * 16 * 10) return;
  int p = t % 10, r = t / 10;
  int b = r / 16, j = r % 16;
  neighP[t] = snp[(b * 17 + order[b * 16 + j]) * 10 + p];
}

// act: 0=none, 1=relu, 2=sigmoid ; optional residual add
__global__ void linear_kernel(const float* __restrict__ X, const float* __restrict__ W,
                              const float* __restrict__ bias, const float* __restrict__ residual,
                              float* __restrict__ Y, int rows, int K, int N, int act) {
  int t = blockIdx.x * blockDim.x + threadIdx.x;
  if (t >= rows * N) return;
  int r = t / N, o = t % N;
  float s = bias ? bias[o] : 0.f;
  for (int k = 0; k < K; ++k) s += X[(size_t)r * K + k] * W[(size_t)k * N + o];
  if (residual) s += residual[t];
  if (act == 1) s = s > 0.f ? s : 0.f;
  else if (act == 2) s = 1.f / (1.f + __expf(-s));
  Y[t] = s;
}

// block per (b,h): softmax over 16 kv, 64 threads
__global__ void attn_kernel(const float* __restrict__ q, const float* __restrict__ k,
                            const float* __restrict__ v, float* __restrict__ o) {
  __shared__ float sc[16];
  int b = blockIdx.x >> 2, h = blockIdx.x & 3;
  int tid = threadIdx.x;
  if (tid < 16) {
    float s = 0.f;
    for (int d = 0; d < 64; ++d)
      s += q[b * HID + h * 64 + d] * k[(b * 16 + tid) * HID + h * 64 + d];
    sc[tid] = s * 0.125f; // 1/sqrt(64)
  }
  __syncthreads();
  if (tid == 0) {
    float m = sc[0];
    for (int j = 1; j < 16; ++j) m = sc[j] > m ? sc[j] : m;
    float sum = 0.f;
    for (int j = 0; j < 16; ++j) { sc[j] = __expf(sc[j] - m); sum += sc[j]; }
    float inv = 1.f / sum;
    for (int j = 0; j < 16; ++j) sc[j] *= inv;
  }
  __syncthreads();
  float acc = 0.f;
  for (int j = 0; j < 16; ++j) acc += sc[j] * v[(b * 16 + j) * HID + h * 64 + tid];
  o[b * HID + h * 64 + tid] = acc;
}

// mr = [max_q f1 (64) | max_q f3 (256) | max_q f4 (512)]
__global__ void mr_kernel(const unsigned short* __restrict__ f1, const unsigned short* __restrict__ f3,
                          const unsigned short* __restrict__ f4, float* __restrict__ mr) {
  int t = blockIdx.x * blockDim.x + threadIdx.x;
  if (t >= B_SZ * 832) return;
  int b = t / 832, c = t % 832;
  float m = -1e30f;
  if (c < 64) {
    for (int qi = 0; qi < 1024; ++qi) { float x = bf2f(f1[((size_t)b * 1024 + qi) * 64 + c]); m = x > m ? x : m; }
  } else if (c < 320) {
    int cc = c - 64;
    for (int qi = 0; qi < 64; ++qi) { float x = bf2f(f3[((size_t)b * 64 + qi) * 256 + cc]); m = x > m ? x : m; }
  } else {
    int cc = c - 320;
    for (int qi = 0; qi < 16; ++qi) { float x = bf2f(f4[((size_t)b * 16 + qi) * 512 + cc]); m = x > m ? x : m; }
  }
  mr[t] = m;
}

__global__ void concat_merged_kernel(const float* __restrict__ selfEmb, const float* __restrict__ comm,
                                     const float* __restrict__ nf, const float* __restrict__ af,
                                     float* __restrict__ merged) {
  int t = blockIdx.x * blockDim.x + threadIdx.x;
  if (t >= B_SZ * 832) return;
  int b = t / 832, c = t % 832;
  float v;
  if (c < 256) v = selfEmb[b * 256 + c];
  else if (c < 512) v = comm[b * 256 + (c - 256)];
  else if (c < 576) v = nf[b * 64 + (c - 512)];
  else v = af[b * 256 + (c - 576)];
  merged[t] = v;
}

// ---------------------------------------------------------------------------
// Host side
// ---------------------------------------------------------------------------
static void run_sa_stage(hipStream_t stream, const float* coordsIn, int Nin,
                         const unsigned short* featsIn, int Cfeat, int npoint, float r2,
                         int Kpad,
                         const unsigned short* W0, const float* b0, int Nc0,
                         const unsigned short* W1, const float* b1, int Nc1,
                         const unsigned short* W2, const float* b2, int Nc2,
                         float* dists, int* fpsIdx, float* newXyz, int* gi,
                         unsigned short* Xar, unsigned short* H1, unsigned short* H2,
                         unsigned short* featsOut) {
  fps_kernel<<<B_SZ, 1024, 0, stream>>>(coordsIn, dists, fpsIdx, Nin, npoint);
  int bq = B_SZ * npoint;
  gather_xyz_kernel<<<CDIV(bq, 128), 128, 0, stream>>>(coordsIn, fpsIdx, newXyz, bq, npoint, Nin);
  ball_query_kernel<<<CDIV(bq, 128), 128, 0, stream>>>(coordsIn, newXyz, gi, bq, npoint, Nin, r2);
  int M = bq * 32;
  int totX = M * Kpad;
  group_kernel<<<CDIV(totX, 256), 256, 0, stream>>>(coordsIn, newXyz, gi, featsIn, Xar,
                                                    totX, npoint, Kpad, Cfeat, Nin);
  gemm_bf16_wmma_kernel<<<dim3(M / 64, Nc0 / 32), 256, 0, stream>>>(Xar, W0, b0, H1, M, Nc0, Kpad);
  gemm_bf16_wmma_kernel<<<dim3(M / 64, Nc1 / 32), 256, 0, stream>>>(H1, W1, b1, H2, M, Nc1, Nc0);
  gemm_bf16_wmma_kernel<<<dim3(M / 64, Nc2 / 32), 256, 0, stream>>>(H2, W2, b2, H1, M, Nc2, Nc1);
  maxpool_kernel<<<CDIV(bq * Nc2, 256), 256, 0, stream>>>(H1, featsOut, bq * Nc2, Nc2);
}

extern "C" void kernel_launch(void* const* d_in, const int* in_sizes, int n_in,
                              void* d_out, int out_size, void* d_ws, size_t ws_size,
                              hipStream_t stream) {
  (void)in_sizes; (void)n_in; (void)out_size; (void)ws_size;

  // ---- inputs (setup_inputs order; params dict flattened key-sorted) ----
  const float* snp     = (const float*)d_in[0];
  const int*   mask    = (const int*)d_in[1];
  const float* neigh   = (const float*)d_in[2];
  const float* actions = (const float*)d_in[3];
  const float* vlp     = (const float*)d_in[4];
  const float* aeW = (const float*)d_in[5],  *aeB = (const float*)d_in[6];
  const float* ceW = (const float*)d_in[7],  *ceB = (const float*)d_in[8];
  const float* nm1W = (const float*)d_in[9], *nm1B = (const float*)d_in[10];
  const float* nm2W = (const float*)d_in[11],*nm2B = (const float*)d_in[12];
  const float* o1W = (const float*)d_in[13], *o1B = (const float*)d_in[14];
  const float* o2W = (const float*)d_in[15], *o2B = (const float*)d_in[16];
  const float* peW = (const float*)d_in[17], *peB = (const float*)d_in[18];
  // sa layers: d_in[19+2i] = W_i, d_in[20+2i] = b_i, i=0..11
  const float* wkW = (const float*)d_in[43], *wkB = (const float*)d_in[44];
  const float* woW = (const float*)d_in[45], *woB = (const float*)d_in[46];
  const float* wqW = (const float*)d_in[47], *wqB = (const float*)d_in[48];
  const float* wvW = (const float*)d_in[49], *wvB = (const float*)d_in[50];

  // ---- workspace arena ----
  char* ws = (char*)d_ws;
  size_t cur = 0;
  auto alloc = [&](size_t bytes) -> void* {
    void* p = ws + cur;
    cur = (cur + bytes + 255) & ~(size_t)255;
    return p;
  };

  float* coords0 = (float*)alloc((size_t)B_SZ * N0_PTS * 3 * 4);
  unsigned short* feats0 = (unsigned short*)alloc((size_t)B_SZ * N0_PTS * 9 * 2);
  float* dists = (float*)alloc((size_t)B_SZ * N0_PTS * 4);
  int* fpsIdx = (int*)alloc((size_t)B_SZ * 1024 * 4);
  float* coords1 = (float*)alloc((size_t)B_SZ * 1024 * 3 * 4);
  float* coords2 = (float*)alloc((size_t)B_SZ * 256 * 3 * 4);
  float* coords3 = (float*)alloc((size_t)B_SZ * 64 * 3 * 4);
  float* coords4 = (float*)alloc((size_t)B_SZ * 16 * 3 * 4);
  int* gi = (int*)alloc((size_t)B_SZ * 1024 * 32 * 4);
  unsigned short* Xar = (unsigned short*)alloc((size_t)B_SZ * 1024 * 32 * 32 * 2); // 8 MB max
  unsigned short* H1 = (unsigned short*)alloc((size_t)B_SZ * 1024 * 32 * 64 * 2);  // 16.8 MB max
  unsigned short* H2 = (unsigned short*)alloc((size_t)B_SZ * 1024 * 32 * 32 * 2);  // 8.4 MB max
  unsigned short* feats1 = (unsigned short*)alloc((size_t)B_SZ * 1024 * 64 * 2);
  unsigned short* feats2 = (unsigned short*)alloc((size_t)B_SZ * 256 * 128 * 2);
  unsigned short* feats3 = (unsigned short*)alloc((size_t)B_SZ * 64 * 256 * 2);
  unsigned short* feats4 = (unsigned short*)alloc((size_t)B_SZ * 16 * 512 * 2);

  static const int saK[12]    = {12, 32, 32, 67, 64, 64, 131, 128, 128, 259, 256, 256};
  static const int saN[12]    = {32, 32, 64, 64, 64, 128, 128, 128, 256, 256, 256, 512};
  static const int saKpad[12] = {32, 32, 32, 96, 64, 64, 160, 128, 128, 288, 256, 256};
  unsigned short* Wbf[12];
  for (int i = 0; i < 12; ++i)
    Wbf[i] = (unsigned short*)alloc((size_t)saKpad[i] * saN[i] * 2);

  int* selfIdx = (int*)alloc(B_SZ * 4);
  int* order = (int*)alloc(B_SZ * 16 * 4);
  float* selfP = (float*)alloc(B_SZ * 10 * 4);
  float* neighP = (float*)alloc(B_SZ * 16 * 10 * 4);
  float* selfEmb = (float*)alloc(B_SZ * HID * 4);
  float* neighEmb = (float*)alloc(B_SZ * 16 * HID * 4);
  float* qbuf = (float*)alloc(B_SZ * HID * 4);
  float* kbuf = (float*)alloc(B_SZ * 16 * HID * 4);
  float* vbuf = (float*)alloc(B_SZ * 16 * HID * 4);
  float* attno = (float*)alloc(B_SZ * HID * 4);
  float* comm = (float*)alloc(B_SZ * HID * 4);
  float* mrbuf = (float*)alloc(B_SZ * 832 * 4);
  float* nf1 = (float*)alloc(B_SZ * HID * 4);
  float* nfbuf = (float*)alloc(B_SZ * 64 * 4);
  float* afbuf = (float*)alloc(B_SZ * HID * 4);
  float* merged = (float*)alloc(B_SZ * 832 * 4);
  float* o1h = (float*)alloc(B_SZ * HID * 4);

  // ---- weight conversion (f32 -> zero-K-padded bf16) ----
  for (int i = 0; i < 12; ++i) {
    int tot = saKpad[i] * saN[i];
    wpad_kernel<<<CDIV(tot, 256), 256, 0, stream>>>(
        (const float*)d_in[19 + 2 * i], Wbf[i], saK[i], saN[i], saKpad[i]);
  }

  // ---- voxel grid -> points ----
  {
    int total = B_SZ * N0_PTS;
    ply2data_kernel<<<CDIV(total, 256), 256, 0, stream>>>(neigh, vlp, coords0, feats0, total);
  }

  // ---- four set-abstraction stages (WMMA GEMM chains) ----
  const float* saB[12];
  for (int i = 0; i < 12; ++i) saB[i] = (const float*)d_in[20 + 2 * i];

  run_sa_stage(stream, coords0, N0_PTS, feats0, 9, 1024, 0.01f, 32,
               Wbf[0], saB[0], 32, Wbf[1], saB[1], 32, Wbf[2], saB[2], 64,
               dists, fpsIdx, coords1, gi, Xar, H1, H2, feats1);
  run_sa_stage(stream, coords1, 1024, feats1, 64, 256, 0.04f, 96,
               Wbf[3], saB[3], 64, Wbf[4], saB[4], 64, Wbf[5], saB[5], 128,
               dists, fpsIdx, coords2, gi, Xar, H1, H2, feats2);
  run_sa_stage(stream, coords2, 256, feats2, 128, 64, 0.16f, 160,
               Wbf[6], saB[6], 128, Wbf[7], saB[7], 128, Wbf[8], saB[8], 256,
               dists, fpsIdx, coords3, gi, Xar, H1, H2, feats3);
  run_sa_stage(stream, coords3, 64, feats3, 256, 16, 0.64f, 288,
               Wbf[9], saB[9], 256, Wbf[10], saB[10], 256, Wbf[11], saB[11], 512,
               dists, fpsIdx, coords4, gi, Xar, H1, H2, feats4);

  // ---- self/neighbor split + embeddings ----
  mask_order_kernel<<<1, B_SZ, 0, stream>>>(mask, selfIdx, order, B_SZ, 17);
  gather_self_kernel<<<1, 64, 0, stream>>>(snp, selfIdx, selfP);
  gather_neigh_kernel<<<CDIV(B_SZ * 16 * 10, 128), 128, 0, stream>>>(snp, order, neighP);
  linear_kernel<<<CDIV(B_SZ * HID, 256), 256, 0, stream>>>(selfP, peW, peB, nullptr, selfEmb, B_SZ, 10, HID, 1);
  linear_kernel<<<CDIV(B_SZ * 16 * HID, 256), 256, 0, stream>>>(neighP, ceW, ceB, nullptr, neighEmb, B_SZ * 16, 10, HID, 0);

  // ---- multi-head attention (4 heads, 16 kv) ----
  linear_kernel<<<CDIV(B_SZ * HID, 256), 256, 0, stream>>>(selfEmb, wqW, wqB, nullptr, qbuf, B_SZ, HID, HID, 0);
  linear_kernel<<<CDIV(B_SZ * 16 * HID, 256), 256, 0, stream>>>(neighEmb, wkW, wkB, nullptr, kbuf, B_SZ * 16, HID, HID, 0);
  linear_kernel<<<CDIV(B_SZ * 16 * HID, 256), 256, 0, stream>>>(neighEmb, wvW, wvB, nullptr, vbuf, B_SZ * 16, HID, HID, 0);
  attn_kernel<<<B_SZ * 4, 64, 0, stream>>>(qbuf, kbuf, vbuf, attno);
  linear_kernel<<<CDIV(B_SZ * HID, 256), 256, 0, stream>>>(attno, woW, woB, selfEmb, comm, B_SZ, HID, HID, 0);

  // ---- global pooled features -> nf ----
  mr_kernel<<<CDIV(B_SZ * 832, 128), 128, 0, stream>>>(feats1, feats3, feats4, mrbuf);
  linear_kernel<<<CDIV(B_SZ * HID, 256), 256, 0, stream>>>(mrbuf, nm1W, nm1B, nullptr, nf1, B_SZ, 832, HID, 1);
  linear_kernel<<<CDIV(B_SZ * 64, 256), 256, 0, stream>>>(nf1, nm2W, nm2B, nullptr, nfbuf, B_SZ, HID, 64, 1);

  // ---- action embedding + output head ----
  linear_kernel<<<CDIV(B_SZ * HID, 256), 256, 0, stream>>>(actions, aeW, aeB, nullptr, afbuf, B_SZ, 15, HID, 1);
  concat_merged_kernel<<<CDIV(B_SZ * 832, 256), 256, 0, stream>>>(selfEmb, comm, nfbuf, afbuf, merged);
  linear_kernel<<<CDIV(B_SZ * HID, 256), 256, 0, stream>>>(merged, o1W, o1B, nullptr, o1h, B_SZ, 832, HID, 1);
  linear_kernel<<<1, 64, 0, stream>>>(o1h, o2W, o2B, nullptr, (float*)d_out, B_SZ, HID, 1, 2);
}
